// FNO2d_43654047597247
// MI455X (gfx1250) — compile-verified, matrix operand order checked by reading, not run on hardware
//
#include <hip/hip_runtime.h>
#include <math.h>

// ---------------------------------------------------------------------------
// FNO2d for MI455X (gfx1250): WMMA f16 (f32 accum) for the dominant bicubic
// resize GEMMs; f32 VALU for the skinny spectral/norm/1x1 pieces.
// All WMMA operands are staged in LDS so each fragment is 2x ds_load_b128;
// all intermediate C tiles are stored transposed => packed b128 stores.
// ---------------------------------------------------------------------------

typedef __attribute__((ext_vector_type(16))) _Float16 v16h;
typedef __attribute__((ext_vector_type(8)))  _Float16 v8h;
typedef __attribute__((ext_vector_type(8)))  float    v8f;

#define Bn   64
#define Sdim 64
#define Rdim 128
#define Cw   20
#define FHn  23   // 2*M1-1
#define FWn  12   // M2

// ---------------------- WMMA helpers (wave32) ------------------------------
// Fragment load where the lane-indexed dimension is the LDS row and K is
// contiguous. Serves both A (lane = M row) and B^T (lane = N col of B, i.e.
// row of the transposed-source matrix). Two 16B ds_load_b128 + shuffle.
__device__ __forceinline__ v16h frag_load_rowK(const _Float16* base, int ld, int row0, int k0) {
    int lane = threadIdx.x & 31;
    int m = lane & 15, hi = (lane >> 4) & 1;
    const _Float16* p = base + (size_t)(row0 + m) * ld + k0 + hi * 8;
    v8h lo = *(const v8h*)p;
    v8h hh = *(const v8h*)(p + 16);
    return __builtin_shufflevector(lo, hh, 0, 1, 2, 3, 4, 5, 6, 7, 8, 9, 10, 11, 12, 13, 14, 15);
}

__device__ __forceinline__ v8f wmma_f16(v16h a, v16h b, v8f c) {
    return __builtin_amdgcn_wmma_f32_16x16x32_f16(false, a, false, b, (short)0, c, false, false);
}

__device__ __forceinline__ float gelu_exact(float x) {
    return 0.5f * x * (1.0f + erff(x * 0.70710678118654752440f));
}

// Store C tile transposed: dst is row-major with leading dim ld, dst row =
// C column (col0+n), dst col = C row (row0+v+hi*8) -> contiguous in v,
// one packed 16B store per lane.
__device__ __forceinline__ void store_T_h16(_Float16* dst, int ld, int col0, int row0, v8f acc) {
    int lane = threadIdx.x & 31;
    int hi = (lane >> 4) & 1, nlo = lane & 15;
    v8h cv;
#pragma unroll
    for (int v = 0; v < 8; ++v) cv[v] = (_Float16)acc[v];
    *(v8h*)(dst + (size_t)(col0 + nlo) * ld + row0 + hi * 8) = cv;
}

__device__ __forceinline__ void store_T_h16_gelu(_Float16* dst, int ld, int col0, int row0, v8f acc) {
    int lane = threadIdx.x & 31;
    int hi = (lane >> 4) & 1, nlo = lane & 15;
    v8h cv;
#pragma unroll
    for (int v = 0; v < 8; ++v) cv[v] = (_Float16)gelu_exact(acc[v]);
    *(v8h*)(dst + (size_t)(col0 + nlo) * ld + row0 + hi * 8) = cv;
}

__device__ __forceinline__ void store_T_f32(float* dst, int ld, int col0, int row0, v8f acc) {
    int lane = threadIdx.x & 31;
    int hi = (lane >> 4) & 1, nlo = lane & 15;
    *(v8f*)(dst + (size_t)(col0 + nlo) * ld + row0 + hi * 8) = acc;
}

// ---------------------- constant-matrix construction -----------------------

__device__ __forceinline__ float cubicf(float x) {
    const float A = -0.75f;
    float ax = fabsf(x);
    if (ax <= 1.0f) return ((A + 2.0f) * ax - (A + 3.0f)) * ax * ax + 1.0f;
    if (ax < 2.0f)  return (((ax - 5.0f) * ax + 8.0f) * ax - 4.0f) * A;
    return 0.0f;
}

__device__ void resize_row(int i, int in_size, int out_size, float* row) {
    for (int j = 0; j < in_size; ++j) row[j] = 0.0f;
    float scale    = (float)in_size / (float)out_size;
    float support  = scale > 1.0f ? 2.0f * scale : 2.0f;
    float invscale = scale > 1.0f ? 1.0f / scale : 1.0f;
    float center   = (i + 0.5f) * scale;
    int xmin = (int)(center - support + 0.5f); if (xmin < 0) xmin = 0;
    int xmax = (int)(center + support + 0.5f); if (xmax > in_size) xmax = in_size;
    float sum = 0.0f;
    for (int j = xmin; j < xmax; ++j) sum += cubicf(((float)j - center + 0.5f) * invscale);
    float inv = 1.0f / sum;
    for (int j = xmin; j < xmax; ++j) row[j] = cubicf(((float)j - center + 0.5f) * invscale) * inv;
}

__global__ void init_consts_kernel(float* MU, float* MD,
                                   float* FHr, float* FHi, float* FWr, float* FWi,
                                   float* GHr, float* GHi, float* GWr, float* GWi) {
    int tid = threadIdx.x;
    for (int i = tid; i < Rdim; i += 256) resize_row(i, Sdim, Rdim, MU + (size_t)i * Sdim);
    for (int i = tid; i < Sdim; i += 256) resize_row(i, Rdim, Sdim, MD + (size_t)i * Rdim);
    const float TP = 6.283185307179586476925f;
    for (int t = tid; t < FHn * Sdim; t += 256) {
        int m = t / Sdim, h = t % Sdim;
        float ang = -TP * (float)(m - 11) * (float)h / 64.0f;
        FHr[t] = cosf(ang); FHi[t] = sinf(ang);
    }
    for (int t = tid; t < FWn * Sdim; t += 256) {
        int n = t / Sdim, w = t % Sdim;
        float ang = -TP * (float)n * (float)w / 64.0f;
        FWr[t] = cosf(ang); FWi[t] = sinf(ang);
    }
    for (int t = tid; t < Sdim * FHn; t += 256) {
        int h = t / FHn, m = t % FHn;
        float ang = TP * (float)(m - 11) * (float)h / 64.0f;
        GHr[t] = cosf(ang); GHi[t] = sinf(ang);
    }
    for (int t = tid; t < FWn * Sdim; t += 256) {
        int n = t / Sdim, w = t % Sdim;
        float ang = TP * (float)n * (float)w / 64.0f;
        float c = (n == 0 ? 1.0f : 2.0f) / 4096.0f;
        GWr[t] = c * cosf(ang); GWi[t] = c * sinf(ang);
    }
}

// ---------------------- elementwise / small kernels ------------------------

__global__ void lifting_kernel(const float* __restrict__ X, const float* __restrict__ pw,
                               const float* __restrict__ pb, float* __restrict__ h) {
    int idx = blockIdx.x * blockDim.x + threadIdx.x;
    if (idx >= Bn * Sdim * Sdim) return;
    int xy = idx & 4095, b = idx >> 12;
    int xx = xy >> 6, yy = xy & 63;
    float v[12];
    const float* xp = X + (size_t)idx * 10;
#pragma unroll
    for (int k = 0; k < 10; ++k) v[k] = xp[k];
    v[10] = (float)xx * (1.0f / 63.0f);
    v[11] = (float)yy * (1.0f / 63.0f);
#pragma unroll
    for (int c = 0; c < Cw; ++c) {
        float acc = pb[c];
#pragma unroll
        for (int k = 0; k < 12; ++k) acc += pw[c * 12 + k] * v[k];
        h[(((size_t)(b * Cw + c)) << 12) + xy] = acc;
    }
}

__global__ __launch_bounds__(256) void instnorm_kernel(const float* __restrict__ in,
                                                       float* __restrict__ out) {
    __shared__ float s1[256], s2[256];
    const float* p = in + ((size_t)blockIdx.x << 12);
    float* q = out + ((size_t)blockIdx.x << 12);
    int tid = threadIdx.x;
    float s = 0.0f, ss = 0.0f;
    for (int i = tid; i < 4096; i += 256) { float v = p[i]; s += v; ss += v * v; }
    s1[tid] = s; s2[tid] = ss;
    __syncthreads();
    for (int off = 128; off > 0; off >>= 1) {
        if (tid < off) { s1[tid] += s1[tid + off]; s2[tid] += s2[tid + off]; }
        __syncthreads();
    }
    float mu  = s1[0] * (1.0f / 4096.0f);
    float var = s2[0] * (1.0f / 4096.0f) - mu * mu;
    float inv = rsqrtf(var + 1e-5f);
    for (int i = tid; i < 4096; i += 256) q[i] = (p[i] - mu) * inv;
}

__global__ void conv1x1_kernel(const float* __restrict__ in, const float* __restrict__ w,
                               const float* __restrict__ bias, const float* __restrict__ add,
                               float* __restrict__ out, int Cin, int Cout, int n) {
    int idx = blockIdx.x * blockDim.x + threadIdx.x;
    if (idx >= n) return;
    int xy = idx & 4095;
    int rest = idx >> 12;
    int o = rest % Cout, b = rest / Cout;
    const float* ip = in + (((size_t)b * Cin) << 12) + xy;
    const float* wp = w + (size_t)o * Cin;
    __builtin_prefetch(ip, 0, 0);   // global_prefetch_b8
    float acc = bias[o];
    for (int i = 0; i < Cin; ++i) acc += ip[(size_t)i << 12] * wp[i];
    if (add) acc += add[idx];
    out[idx] = acc;
}

// ---------------------- spectral conv (truncated DFT) ----------------------

__global__ void ffth_kernel(const float* __restrict__ x, const float* __restrict__ FHr,
                            const float* __restrict__ FHi, float* __restrict__ T1r,
                            float* __restrict__ T1i) {
    int idx = blockIdx.x * blockDim.x + threadIdx.x;
    const int n = Bn * Cw * FHn * Sdim;
    if (idx >= n) return;
    int w = idx & 63;
    int m = (idx >> 6) % FHn;
    int bi = idx / (FHn * Sdim);
    const float* p  = x + (((size_t)bi) << 12) + w;
    const float* cr = FHr + m * Sdim;
    const float* ci = FHi + m * Sdim;
    float ar = 0.0f, ai = 0.0f;
    for (int h = 0; h < 64; ++h) { float v = p[h << 6]; ar += cr[h] * v; ai += ci[h] * v; }
    T1r[idx] = ar; T1i[idx] = ai;
}

__global__ void fftw_kernel(const float* __restrict__ T1r, const float* __restrict__ T1i,
                            const float* __restrict__ FWr, const float* __restrict__ FWi,
                            float* __restrict__ XSr, float* __restrict__ XSi) {
    int idx = blockIdx.x * blockDim.x + threadIdx.x;
    const int n = Bn * Cw * FHn * FWn;
    if (idx >= n) return;
    int nn = idx % FWn;
    int m  = (idx / FWn) % FHn;
    int bi = idx / (FHn * FWn);
    const float* tr = T1r + ((size_t)bi * FHn + m) * Sdim;
    const float* ti = T1i + ((size_t)bi * FHn + m) * Sdim;
    const float* cr = FWr + nn * Sdim;
    const float* ci = FWi + nn * Sdim;
    float ar = 0.0f, ai = 0.0f;
    for (int w = 0; w < 64; ++w) {
        float a = tr[w], b = ti[w], c = cr[w], d = ci[w];
        ar += a * c - b * d;
        ai += a * d + b * c;
    }
    XSr[idx] = ar; XSi[idx] = ai;
}

__global__ void modemix_kernel(const float* __restrict__ XSr, const float* __restrict__ XSi,
                               const float* __restrict__ y0r, const float* __restrict__ y0i,
                               const float* __restrict__ c00, const float* __restrict__ yxr,
                               const float* __restrict__ yxi, float* __restrict__ OUr,
                               float* __restrict__ OUi) {
    int idx = blockIdx.x * blockDim.x + threadIdx.x;
    const int n = Bn * Cw * FHn * FWn;
    if (idx >= n) return;
    int nn = idx % FWn;
    int m  = (idx / FWn) % FHn;
    int o  = (idx / (FHn * FWn)) % Cw;
    int b  = idx / (Cw * FHn * FWn);
    float ar = 0.0f, ai = 0.0f;
    for (int i = 0; i < Cw; ++i) {
        size_t xoff = (((size_t)(b * Cw + i) * FHn) + m) * FWn + nn;
        float xr = XSr[xoff], xi = XSi[xoff];
        int io = i * Cw + o;
        float wr, wi;
        if (nn == 0) {
            if (m < 11)       { wr = y0r[io * 11 + m];        wi = y0i[io * 11 + m]; }
            else if (m == 11) { wr = c00[io];                 wi = 0.0f; }
            else              { wr = y0r[io * 11 + (22 - m)]; wi = -y0i[io * 11 + (22 - m)]; }
        } else {
            size_t wo = (size_t)io * (FHn * 11) + m * 11 + (nn - 1);
            wr = yxr[wo]; wi = yxi[wo];
        }
        ar += xr * wr - xi * wi;
        ai += xr * wi + xi * wr;
    }
    OUr[idx] = ar; OUi[idx] = ai;
}

__global__ void invh_kernel(const float* __restrict__ OUr, const float* __restrict__ OUi,
                            const float* __restrict__ GHr, const float* __restrict__ GHi,
                            float* __restrict__ Ur, float* __restrict__ Ui) {
    int idx = blockIdx.x * blockDim.x + threadIdx.x;
    const int n = Bn * Cw * Sdim * FWn;
    if (idx >= n) return;
    int nn = idx % FWn;
    int h  = (idx / FWn) % Sdim;
    int bo = idx / (Sdim * FWn);
    const float* cr = GHr + h * FHn;
    const float* ci = GHi + h * FHn;
    float ar = 0.0f, ai = 0.0f;
    for (int m = 0; m < FHn; ++m) {
        size_t xoff = (size_t)bo * (FHn * FWn) + m * FWn + nn;
        float xr = OUr[xoff], xi = OUi[xoff];
        float c = cr[m], d = ci[m];
        ar += c * xr - d * xi;
        ai += c * xi + d * xr;
    }
    Ur[idx] = ar; Ui[idx] = ai;
}

__global__ void invw_kernel(const float* __restrict__ Ur, const float* __restrict__ Ui,
                            const float* __restrict__ GWr, const float* __restrict__ GWi,
                            float* __restrict__ out) {
    int idx = blockIdx.x * blockDim.x + threadIdx.x;
    const int n = Bn * Cw * Sdim * Sdim;
    if (idx >= n) return;
    int w  = idx & 63;
    int h  = (idx >> 6) & 63;
    int bo = idx >> 12;
    const float* ur = Ur + ((size_t)bo * Sdim + h) * FWn;
    const float* ui = Ui + ((size_t)bo * Sdim + h) * FWn;
    float acc = 0.0f;
#pragma unroll
    for (int nn = 0; nn < FWn; ++nn)
        acc += ur[nn] * GWr[nn * Sdim + w] - ui[nn] * GWi[nn * Sdim + w];
    out[idx] = acc;
}

// ---------------------- fused ud_act: the WMMA workhorse -------------------
// One workgroup (8 wave32 waves) per 64x64 image, in place.
// Each GEMM is arranged so both operands are lane-row/contiguous-K in LDS
// (2x ds_load_b128 per fragment) and each result tile stores packed:
//   mm1: t1^T = x^T * MU^T      -> sT1   (t1 row-major 128x64)
//   mm2: t2   = t1 * MU^T, gelu -> sT2T  (t2^T row-major 128x128)
//   mm3: d1^T = t2^T * MD^T     -> sD1   (d1 row-major 64x128)
//   mm4: out^T = MD * d1^T      -> sOut  (out row-major 64x64, f32)

__global__ __launch_bounds__(256) void ud_act_kernel(float* __restrict__ data,
                                                     const float* __restrict__ MUf,
                                                     const float* __restrict__ MDf) {
    __shared__ _Float16 sMU [Rdim * Sdim];   // 16 KB, MU row-major 128x64
    __shared__ _Float16 sMD [Sdim * Rdim];   // 16 KB, MD row-major 64x128
    __shared__ _Float16 sXT [Sdim * Sdim];   //  8 KB, x^T row-major
    __shared__ _Float16 sT1 [Rdim * Sdim];   // 16 KB, t1 row-major
    __shared__ _Float16 sT2T[Rdim * Rdim];   // 32 KB, t2^T row-major
    __shared__ _Float16 sD1 [Sdim * Rdim];   // 16 KB, d1 row-major
    __shared__ float    sOut[Sdim * Sdim];   // 16 KB, out row-major (f32)

    float* img = data + ((size_t)blockIdx.x << 12);
    int tid = threadIdx.x;
    for (int i = tid; i < Rdim * Sdim; i += 256) { sMU[i] = (_Float16)MUf[i]; sMD[i] = (_Float16)MDf[i]; }
    for (int i = tid; i < Sdim * Sdim; i += 256) {
        int r = i >> 6, c = i & 63;
        sXT[c * Sdim + r] = (_Float16)img[i];
    }
    __syncthreads();

    int w = tid >> 5;

    { // mm1: t1^T (64x128) = x^T (64x64) * MU^T (64x128), K=64
        int row0 = (w & 3) * 16;
        v16h a0 = frag_load_rowK(sXT, Sdim, row0, 0);
        v16h a1 = frag_load_rowK(sXT, Sdim, row0, 32);
#pragma unroll
        for (int j = 0; j < 4; ++j) {
            int col0 = ((w >> 2) * 4 + j) * 16;
            v8f acc = {};
            acc = wmma_f16(a0, frag_load_rowK(sMU, Sdim, col0, 0),  acc);
            acc = wmma_f16(a1, frag_load_rowK(sMU, Sdim, col0, 32), acc);
            store_T_h16(sT1, Sdim, col0, row0, acc);   // sT1[t1row][t1col]
        }
    }
    __syncthreads();

    { // mm2: t2 (128x128) = t1 (128x64) * MU^T (64x128), K=64; gelu; store t2^T
        int row0 = w * 16;
        v16h a0 = frag_load_rowK(sT1, Sdim, row0, 0);
        v16h a1 = frag_load_rowK(sT1, Sdim, row0, 32);
#pragma unroll
        for (int ct = 0; ct < 8; ++ct) {
            int col0 = ct * 16;
            v8f acc = {};
            acc = wmma_f16(a0, frag_load_rowK(sMU, Sdim, col0, 0),  acc);
            acc = wmma_f16(a1, frag_load_rowK(sMU, Sdim, col0, 32), acc);
            store_T_h16_gelu(sT2T, Rdim, col0, row0, acc);
        }
    }
    __syncthreads();

    { // mm3: d1^T (128x64) = t2^T (128x128) * MD^T (128x64), K=128
        int row0 = w * 16;
        v16h aK[4];
#pragma unroll
        for (int kt = 0; kt < 4; ++kt) aK[kt] = frag_load_rowK(sT2T, Rdim, row0, kt * 32);
#pragma unroll
        for (int j = 0; j < 4; ++j) {
            int col0 = j * 16;
            v8f acc = {};
#pragma unroll
            for (int kt = 0; kt < 4; ++kt)
                acc = wmma_f16(aK[kt], frag_load_rowK(sMD, Rdim, col0, kt * 32), acc);
            store_T_h16(sD1, Rdim, col0, row0, acc);   // sD1[d1row][d1col]
        }
    }
    __syncthreads();

    { // mm4: out^T (64x64) = MD (64x128) * d1^T (128x64), K=128
        int row0 = (w & 3) * 16;
        v16h aK[4];
#pragma unroll
        for (int kt = 0; kt < 4; ++kt) aK[kt] = frag_load_rowK(sMD, Rdim, row0, kt * 32);
#pragma unroll
        for (int j = 0; j < 2; ++j) {
            int col0 = ((w >> 2) * 2 + j) * 16;
            v8f acc = {};
#pragma unroll
            for (int kt = 0; kt < 4; ++kt)
                acc = wmma_f16(aK[kt], frag_load_rowK(sD1, Rdim, col0, kt * 32), acc);
            store_T_f32(sOut, Sdim, col0, row0, acc);  // sOut[outrow][outcol]
        }
    }
    __syncthreads();

    // coalesced copy to global
    for (int i = tid; i < 1024; i += 256)
        ((float4*)img)[i] = ((const float4*)sOut)[i];
}

// ---------------------------------------------------------------------------

extern "C" void kernel_launch(void* const* d_in, const int* in_sizes, int n_in,
                              void* d_out, int out_size, void* d_ws, size_t ws_size,
                              hipStream_t stream) {
    (void)in_sizes; (void)n_in; (void)out_size; (void)ws_size;

    const float* X    = (const float*)d_in[0];
    const float* p_w  = (const float*)d_in[1];
    const float* p_b  = (const float*)d_in[2];
    const float* y0r  = (const float*)d_in[3];
    const float* y0i  = (const float*)d_in[4];
    const float* c00  = (const float*)d_in[5];
    const float* yxr  = (const float*)d_in[6];
    const float* yxi  = (const float*)d_in[7];
    const float* maw  = (const float*)d_in[8];
    const float* mab  = (const float*)d_in[9];
    const float* mbw  = (const float*)d_in[10];
    const float* mbb  = (const float*)d_in[11];
    const float* ww   = (const float*)d_in[12];
    const float* wb   = (const float*)d_in[13];
    const float* q1w  = (const float*)d_in[14];
    const float* q1b  = (const float*)d_in[15];
    const float* q2w  = (const float*)d_in[16];
    const float* q2b  = (const float*)d_in[17];

    float* ws = (float*)d_ws;
    size_t off = 0;
    auto alloc = [&](size_t n) { float* p = ws + off; off += n; return p; };

    float* MU  = alloc(Rdim * Sdim);
    float* MD  = alloc(Sdim * Rdim);
    float* FHr = alloc(FHn * Sdim); float* FHi = alloc(FHn * Sdim);
    float* FWr = alloc(FWn * Sdim); float* FWi = alloc(FWn * Sdim);
    float* GHr = alloc(Sdim * FHn); float* GHi = alloc(Sdim * FHn);
    float* GWr = alloc(FWn * Sdim); float* GWi = alloc(FWn * Sdim);

    const size_t IMG   = (size_t)Sdim * Sdim;           // 4096
    const size_t ACT20 = (size_t)Bn * Cw * IMG;         // 5,242,880
    float* bufA = alloc(ACT20);
    float* bufB = alloc(ACT20);
    float* bufC = alloc(ACT20);
    float* hbig = alloc((size_t)Bn * 4 * Cw * IMG);     // 20,971,520

    float* T1r = alloc((size_t)Bn * Cw * FHn * Sdim); float* T1i = alloc((size_t)Bn * Cw * FHn * Sdim);
    float* XSr = alloc((size_t)Bn * Cw * FHn * FWn);  float* XSi = alloc((size_t)Bn * Cw * FHn * FWn);
    float* OUr = alloc((size_t)Bn * Cw * FHn * FWn);  float* OUi = alloc((size_t)Bn * Cw * FHn * FWn);
    float* Ur  = alloc((size_t)Bn * Cw * Sdim * FWn); float* Ui  = alloc((size_t)Bn * Cw * Sdim * FWn);

    auto cdiv = [](int a, int b) { return (a + b - 1) / b; };

    init_consts_kernel<<<1, 256, 0, stream>>>(MU, MD, FHr, FHi, FWr, FWi, GHr, GHi, GWr, GWi);
    lifting_kernel<<<cdiv(Bn * (int)IMG, 256), 256, 0, stream>>>(X, p_w, p_b, bufA);

    float* h  = bufA;
    float* t1 = bufB;
    float* t2 = bufC;

    const int nFH  = Bn * Cw * FHn * Sdim;
    const int nFW  = Bn * Cw * FHn * FWn;
    const int nIH  = Bn * Cw * Sdim * FWn;
    const int nACT = Bn * Cw * (int)IMG;

    for (int l = 0; l < 4; ++l) {
        instnorm_kernel<<<Bn * Cw, 256, 0, stream>>>(h, t1);
        ffth_kernel<<<cdiv(nFH, 256), 256, 0, stream>>>(t1, FHr, FHi, T1r, T1i);
        fftw_kernel<<<cdiv(nFW, 256), 256, 0, stream>>>(T1r, T1i, FWr, FWi, XSr, XSi);
        modemix_kernel<<<cdiv(nFW, 256), 256, 0, stream>>>(
            XSr, XSi,
            y0r + (size_t)l * Cw * Cw * 11, y0i + (size_t)l * Cw * Cw * 11,
            c00 + (size_t)l * Cw * Cw,
            yxr + (size_t)l * Cw * Cw * FHn * 11, yxi + (size_t)l * Cw * Cw * FHn * 11,
            OUr, OUi);
        invh_kernel<<<cdiv(nIH, 256), 256, 0, stream>>>(OUr, OUi, GHr, GHi, Ur, Ui);
        invw_kernel<<<cdiv(nACT, 256), 256, 0, stream>>>(Ur, Ui, GWr, GWi, t2);
        instnorm_kernel<<<Bn * Cw, 256, 0, stream>>>(t2, t1);
        conv1x1_kernel<<<cdiv(nACT, 256), 256, 0, stream>>>(
            t1, maw + (size_t)l * Cw * Cw, mab + (size_t)l * Cw, nullptr, t2, Cw, Cw, nACT);
        ud_act_kernel<<<Bn * Cw, 256, 0, stream>>>(t2, MU, MD);
        conv1x1_kernel<<<cdiv(nACT, 256), 256, 0, stream>>>(
            t2, mbw + (size_t)l * Cw * Cw, mbb + (size_t)l * Cw, nullptr, t1, Cw, Cw, nACT);
        conv1x1_kernel<<<cdiv(nACT, 256), 256, 0, stream>>>(
            h, ww + (size_t)l * Cw * Cw, wb + (size_t)l * Cw, t1, t2, Cw, Cw, nACT);
        if (l < 3) ud_act_kernel<<<Bn * Cw, 256, 0, stream>>>(t2, MU, MD);
        float* old = h; h = t2; t2 = old;
    }

    const int nBIG = Bn * 4 * Cw * (int)IMG;
    conv1x1_kernel<<<cdiv(nBIG, 256), 256, 0, stream>>>(h, q1w, q1b, nullptr, hbig, Cw, 4 * Cw, nBIG);
    ud_act_kernel<<<Bn * 4 * Cw, 256, 0, stream>>>(hbig, MU, MD);
    const int nOUT = Bn * (int)IMG;
    conv1x1_kernel<<<cdiv(nOUT, 256), 256, 0, stream>>>(hbig, q2w, q2b, nullptr, (float*)d_out, 4 * Cw, 1, nOUT);
}
